// CrossAttention_62732292325569
// MI455X (gfx1250) — matrix operand order repaired and checked by reading, not compiled, not run on hardware
//
#include <hip/hip_runtime.h>

// ---------------------------------------------------------------------------
// CrossAttention for MI455X (gfx1250), wave32, WMMA bf16 16x16x32.
// Shapes fixed per reference: B=4, N=M=2048, C=1024, H=16, D=64.
// ---------------------------------------------------------------------------

typedef __attribute__((ext_vector_type(16))) __bf16 v16bf;
typedef __attribute__((ext_vector_type(8)))  __bf16 v8bf;
typedef __attribute__((ext_vector_type(8)))  float  v8f;

#define LSTR 40   // LDS row stride in bf16 elements (80B, 16B-aligned, pads banks)

__device__ __forceinline__ v8f vzero8() {
  v8f v;
#pragma unroll
  for (int i = 0; i < 8; ++i) v[i] = 0.0f;
  return v;
}

__device__ __forceinline__ v16bf cat16(v8bf lo, v8bf hi) {
  return __builtin_shufflevector(lo, hi, 0,1,2,3,4,5,6,7,8,9,10,11,12,13,14,15);
}

// ---- staging helpers: global -> regs, regs -> LDS (with f32->bf16 cvt) ----
template<int MODE>
__device__ __forceinline__ void stage_load(
    const float* __restrict__ Af32, const __bf16* __restrict__ Abf,
    const float* __restrict__ Ww, int rowBase, int colBase, int K, int k0,
    int tid, float4* ra, v8bf* rab, float4* rb)
{
  if constexpr (MODE == 0) {
#pragma unroll
    for (int i = 0; i < 2; ++i) {
      int c = tid + i * 256, r = c >> 2, kc = (c & 3) * 8;
      rab[i] = *(const v8bf*)&Abf[(size_t)(rowBase + r) * K + k0 + kc];
    }
  } else {
#pragma unroll
    for (int i = 0; i < 4; ++i) {
      int c = tid + i * 256, r = c >> 3, kc = (c & 7) * 4;
      ra[i] = *(const float4*)&Af32[(size_t)(rowBase + r) * K + k0 + kc];
    }
  }
#pragma unroll
  for (int i = 0; i < 4; ++i) {
    int c = tid + i * 256, r = c >> 3, kc = (c & 7) * 4;
    rb[i] = *(const float4*)&Ww[(size_t)(colBase + r) * K + k0 + kc];
  }
}

template<int MODE>
__device__ __forceinline__ void stage_store(
    __bf16* Al, __bf16* Bl, int tid,
    const float4* ra, const v8bf* rab, const float4* rb)
{
  if constexpr (MODE == 0) {
#pragma unroll
    for (int i = 0; i < 2; ++i) {
      int c = tid + i * 256, r = c >> 2, kc = (c & 3) * 8;
      *(v8bf*)&Al[r * LSTR + kc] = rab[i];
    }
  } else {
#pragma unroll
    for (int i = 0; i < 4; ++i) {
      int c = tid + i * 256, r = c >> 3, kc = (c & 7) * 4;
      __bf16* p = &Al[r * LSTR + kc];
      p[0] = (__bf16)ra[i].x; p[1] = (__bf16)ra[i].y;
      p[2] = (__bf16)ra[i].z; p[3] = (__bf16)ra[i].w;
    }
  }
#pragma unroll
  for (int i = 0; i < 4; ++i) {
    int c = tid + i * 256, r = c >> 3, kc = (c & 7) * 4;
    __bf16* p = &Bl[r * LSTR + kc];
    p[0] = (__bf16)rb[i].x; p[1] = (__bf16)rb[i].y;
    p[2] = (__bf16)rb[i].z; p[3] = (__bf16)rb[i].w;
  }
}

// ---------------------------------------------------------------------------
// WMMA GEMM:  out[m][n] = sum_k A[m][k] * W[n][k]   (W row-major NxK)
// MODE 0: A bf16, out = f32 (+bias)                       [final projection]
// MODE 1: A f32,  epilogue = RoPE + scale -> qbf (B,H,N,D)       [Q proj]
// MODE 2: A f32,  epilogue = RoPE->kbf (B,H,M,D), V->vbf (B,H,D,M) [KV proj]
// 256 thr = 8 waves (2x4); block tile 128x128; wave tile 64x32; BK=32;
// double-buffered LDS, one barrier per K step, loads hidden behind WMMA.
// ---------------------------------------------------------------------------
template<int MODE>
__global__ __launch_bounds__(256) void wmma_gemm_kernel(
    const float* __restrict__ Af32,
    const __bf16* __restrict__ Abf,
    const float* __restrict__ Ww,
    const float* __restrict__ bias,
    const float* __restrict__ freqs,     // (2048, 32, 2) f32
    float*  __restrict__ outF,
    __bf16* __restrict__ outQ,
    __bf16* __restrict__ outK,
    __bf16* __restrict__ outV,
    int Nout, int K)
{
  __shared__ __bf16 Alds[2][128 * LSTR];
  __shared__ __bf16 Blds[2][128 * LSTR];

  const int tid  = threadIdx.x;
  const int lane = tid & 31;
  const int wid  = tid >> 5;
  const int wm   = wid >> 2;          // 0..1
  const int wn   = wid & 3;           // 0..3
  const int half = lane >> 4;         // 0/1
  const int l16  = lane & 15;
  const int ka   = half * 16;         // A-frag K chunk start
  const int kb   = half * 16;         // B-frag K chunk start
  const int rowBase = blockIdx.y * 128;
  const int colBase = blockIdx.x * 128;

  v8f acc[4][2];
#pragma unroll
  for (int mt = 0; mt < 4; ++mt)
#pragma unroll
    for (int nt = 0; nt < 2; ++nt) acc[mt][nt] = vzero8();

  float4 ra[4], rb[4];
  v8bf   rab[2];

  // prologue: stage tile 0
  stage_load<MODE>(Af32, Abf, Ww, rowBase, colBase, K, 0, tid, ra, rab, rb);
  stage_store<MODE>(Alds[0], Blds[0], tid, ra, rab, rb);
  __syncthreads();

  int buf = 0;
  for (int k0 = 0; k0 < K; k0 += 32) {
    const bool hasNext = (k0 + 32 < K);
    if (hasNext) {
      stage_load<MODE>(Af32, Abf, Ww, rowBase, colBase, K, k0 + 32, tid, ra, rab, rb);
      if (k0 + 64 < K) {   // L2 prefetch two tiles ahead
        __builtin_prefetch(&Ww[(size_t)(colBase + (tid >> 1)) * K + k0 + 64 + (tid & 1) * 16], 0, 1);
        if constexpr (MODE == 0)
          __builtin_prefetch(&Abf[(size_t)(rowBase + (tid >> 1)) * K + k0 + 64], 0, 1);
        else
          __builtin_prefetch(&Af32[(size_t)(rowBase + (tid >> 1)) * K + k0 + 64 + (tid & 1) * 16], 0, 1);
      }
    }

    // ---- fragments from current buffer & 8 WMMAs (loads in flight) ----
    v16bf af[4], bfr[2];
#pragma unroll
    for (int mt = 0; mt < 4; ++mt) {
      const __bf16* base = &Alds[buf][(wm * 64 + mt * 16 + l16) * LSTR + ka];
      af[mt] = cat16(*(const v8bf*)base, *(const v8bf*)(base + 8));
    }
#pragma unroll
    for (int nt = 0; nt < 2; ++nt) {
      const __bf16* base = &Blds[buf][(wn * 32 + nt * 16 + l16) * LSTR + kb];
      bfr[nt] = cat16(*(const v8bf*)base, *(const v8bf*)(base + 8));
    }
#pragma unroll
    for (int mt = 0; mt < 4; ++mt)
#pragma unroll
      for (int nt = 0; nt < 2; ++nt)
        acc[mt][nt] = __builtin_amdgcn_wmma_f32_16x16x32_bf16(
            false, af[mt], false, bfr[nt], (short)0, acc[mt][nt], false, false);

    if (hasNext)
      stage_store<MODE>(Alds[buf ^ 1], Blds[buf ^ 1], tid, ra, rab, rb);
    __syncthreads();
    buf ^= 1;
  }

  // ---- epilogue ----
#pragma unroll
  for (int mt = 0; mt < 4; ++mt) {
#pragma unroll
    for (int nt = 0; nt < 2; ++nt) {
#pragma unroll
      for (int r = 0; r < 8; ++r) {
        const int grow = rowBase + wm * 64 + mt * 16 + r + half * 8;
        const int gcol = colBase + wn * 32 + nt * 16 + l16;
        float val = acc[mt][nt][r];
        if constexpr (MODE == 0) {
          outF[(size_t)grow * Nout + gcol] = val + bias[gcol];
        } else if constexpr (MODE == 1) {
          const int b = grow >> 11, n = grow & 2047;     // N = 2048
          const int h = gcol >> 6,  d = gcol & 63;       // D = 64
          const float partner = __shfl_xor(val, 1, 32);  // RoPE pair = adjacent lane
          const float2 f = *(const float2*)&freqs[((size_t)n * 32 + (d >> 1)) * 2];
          float res = (d & 1) ? (val * f.x + partner * f.y)
                              : (val * f.x - partner * f.y);
          res *= 0.125f;                                 // scale = D^-0.5
          outQ[(((size_t)b * 16 + h) * 2048 + n) * 64 + d] = (__bf16)res;
        } else {
          const int b = grow >> 11, m = grow & 2047;     // M = 2048
          if (gcol < 1024) {                             // K half: RoPE, no scale
            const int h = gcol >> 6, d = gcol & 63;
            const float partner = __shfl_xor(val, 1, 32);
            const float2 f = *(const float2*)&freqs[((size_t)m * 32 + (d >> 1)) * 2];
            float res = (d & 1) ? (val * f.x + partner * f.y)
                                : (val * f.x - partner * f.y);
            outK[(((size_t)b * 16 + h) * 2048 + m) * 64 + d] = (__bf16)res;
          } else {                                       // V half: store transposed
            const int c2 = gcol - 1024;
            const int h = c2 >> 6, d = c2 & 63;
            outV[(((size_t)b * 16 + h) * 64 + d) * 2048 + m] = (__bf16)val;
          }
        }
      }
    }
  }
}

// ---------------------------------------------------------------------------
// Flash attention: 256 thr = 8 waves; each wave owns a 16-row Q tile and runs
// fully decoupled (no cross-wave barriers; per-wave LDS slice, intra-wave
// ordering via in-order LDS + explicit s_wait_dscnt).
// q: (B,H,N,D) bf16 (pre-scaled), k: (B,H,M,D), v: (B,H,D,M) transposed.
// ---------------------------------------------------------------------------
__global__ __launch_bounds__(256) void attn_kernel(
    const __bf16* __restrict__ qbf,
    const __bf16* __restrict__ kbf,
    const __bf16* __restrict__ vbf,
    __bf16* __restrict__ attn)
{
  __shared__ __bf16 pbuf[8][16 * LSTR];

  const int lane = threadIdx.x & 31;
  const int wid  = threadIdx.x >> 5;
  const int half = lane >> 4;
  const int l16  = lane & 15;
  const int b = blockIdx.z, h = blockIdx.y;
  const int qrow = blockIdx.x * 128 + wid * 16;
  const size_t headRow = ((size_t)b * 16 + h) * 2048;        // q/k row base
  const size_t vHead   = ((size_t)b * 16 + h) * 64;          // v (D,M) row base

  // Q fragments: two K-steps over D=64
  v16bf aq[2];
  {
    const __bf16* qp = qbf + (headRow + qrow + l16) * 64;
#pragma unroll
    for (int s = 0; s < 2; ++s) {
      const int kc = s * 32 + half * 16;
      aq[s] = cat16(*(const v8bf*)(qp + kc), *(const v8bf*)(qp + kc + 8));
    }
  }

  v8f o[4];
#pragma unroll
  for (int j = 0; j < 4; ++j) o[j] = vzero8();
  float mi[8], li[8];
#pragma unroll
  for (int r = 0; r < 8; ++r) { mi[r] = -INFINITY; li[r] = 0.0f; }

  for (int m0 = 0; m0 < 2048; m0 += 32) {
    if (m0 + 32 < 2048) {  // prefetch next key/value tiles into cache
      __builtin_prefetch(kbf + (headRow + m0 + 32 + l16) * 64, 0, 1);
      __builtin_prefetch(vbf + (vHead + l16) * 2048 + m0 + 32, 0, 1);
      __builtin_prefetch(vbf + (vHead + 16 + l16) * 2048 + m0 + 32, 0, 1);
    }

    // ---- scores S = q k^T for 32 keys (two 16-key tiles) ----
    v8f sA[2];
#pragma unroll
    for (int t = 0; t < 2; ++t) {
      sA[t] = vzero8();
      const __bf16* kp = kbf + (headRow + m0 + t * 16 + l16) * 64;
#pragma unroll
      for (int s = 0; s < 2; ++s) {
        const int kc = s * 32 + half * 16;
        v16bf bk = cat16(*(const v8bf*)(kp + kc), *(const v8bf*)(kp + kc + 8));
        sA[t] = __builtin_amdgcn_wmma_f32_16x16x32_bf16(
            false, aq[s], false, bk, (short)0, sA[t], false, false);
      }
    }

    // ---- online softmax per row (row = r + half*8, spread over 16 lanes) ----
#pragma unroll
    for (int r = 0; r < 8; ++r) {
      const float v0 = sA[0][r], v1 = sA[1][r];
      float mx = fmaxf(v0, v1);
#pragma unroll
      for (int msk = 1; msk < 16; msk <<= 1)
        mx = fmaxf(mx, __shfl_xor(mx, msk, 32));
      const float mnew = fmaxf(mi[r], mx);
      const float al = __expf(mi[r] - mnew);
      const float e0 = __expf(v0 - mnew);
      const float e1 = __expf(v1 - mnew);
      float sum = e0 + e1;
#pragma unroll
      for (int msk = 1; msk < 16; msk <<= 1)
        sum += __shfl_xor(sum, msk, 32);
      li[r] = li[r] * al + sum;
      mi[r] = mnew;
#pragma unroll
      for (int j = 0; j < 4; ++j) o[j][r] *= al;
      __bf16* pr = &pbuf[wid][(r + half * 8) * LSTR];
      pr[l16]      = (__bf16)e0;
      pr[16 + l16] = (__bf16)e1;
    }

    // intra-wave LDS RAW fence: per-wave pbuf slice, LDS is in-order per wave
    asm volatile("s_wait_dscnt 0" ::: "memory");

    // ---- P (16x32) redistributed into A-fragment layout ----
    const __bf16* pb = &pbuf[wid][l16 * LSTR + half * 16];
    v16bf ap = cat16(*(const v8bf*)pb, *(const v8bf*)(pb + 8));

    // ---- O += P @ V  (V transposed: contiguous along keys) ----
#pragma unroll
    for (int j = 0; j < 4; ++j) {
      const __bf16* vp = vbf + (vHead + j * 16 + l16) * 2048 + m0 + half * 16;
      v16bf bv = cat16(*(const v8bf*)vp, *(const v8bf*)(vp + 8));
      o[j] = __builtin_amdgcn_wmma_f32_16x16x32_bf16(
          false, ap, false, bv, (short)0, o[j], false, false);
    }
    // compiler fence: keep next iteration's P stores after this read
    asm volatile("" ::: "memory");
  }

  // ---- normalize and store (B,N,H*D) bf16 ----
#pragma unroll
  for (int r = 0; r < 8; ++r) {
    const float inv = __frcp_rn(li[r]);
    const int n = qrow + r + half * 8;
#pragma unroll
    for (int j = 0; j < 4; ++j) {
      const int col = h * 64 + j * 16 + l16;
      attn[((size_t)b * 2048 + n) * 1024 + col] = (__bf16)(o[j][r] * inv);
    }
  }
}

// ---------------------------------------------------------------------------
extern "C" void kernel_launch(void* const* d_in, const int* in_sizes, int n_in,
                              void* d_out, int out_size, void* d_ws, size_t ws_size,
                              hipStream_t stream) {
  const float* x     = (const float*)d_in[0];   // (4, 2048, 1024)
  const float* ctx   = (const float*)d_in[1];   // (4, 2048, 1024)
  const float* freqs = (const float*)d_in[2];   // (2048, 32, 2)
  const float* Wq    = (const float*)d_in[3];   // (1024, 1024)
  const float* Wkv   = (const float*)d_in[4];   // (2048, 1024)
  const float* Wproj = (const float*)d_in[5];   // (1024, 1024)
  const float* bproj = (const float*)d_in[6];   // (1024,)
  float* out = (float*)d_out;

  const size_t E = (size_t)8192 * 1024;         // elements per bf16 buffer
  __bf16* qb = (__bf16*)d_ws;
  __bf16* kb = qb + E;
  __bf16* vb = kb + E;
  __bf16* at = vb + E;

  // 1) Q = x @ Wq^T, fused RoPE + scale -> qb (B,H,N,D)
  wmma_gemm_kernel<1><<<dim3(8, 64), 256, 0, stream>>>(
      x, nullptr, Wq, nullptr, freqs, nullptr, qb, nullptr, nullptr, 1024, 1024);

  // 2) KV = ctx @ Wkv^T, fused RoPE on K -> kb; V transposed -> vb
  wmma_gemm_kernel<2><<<dim3(16, 64), 256, 0, stream>>>(
      ctx, nullptr, Wkv, nullptr, freqs, nullptr, nullptr, kb, vb, 2048, 1024);

  // 3) flash attention -> at (B*N, C) bf16
  attn_kernel<<<dim3(16, 16, 4), 256, 0, stream>>>(qb, kb, vb, at);

  // 4) out = at @ Wproj^T + bproj  (fp32 output)
  wmma_gemm_kernel<0><<<dim3(8, 64), 256, 0, stream>>>(
      nullptr, at, Wproj, bproj, nullptr, out, nullptr, nullptr, nullptr, 1024, 1024);
}
